// QLinearLR_73864847556829
// MI455X (gfx1250) — matrix operand order repaired and checked by reading, not compiled, not run on hardware
//
#include <hip/hip_runtime.h>

typedef __bf16 bf16;
typedef __attribute__((ext_vector_type(16))) __bf16 v16bf;
typedef __attribute__((ext_vector_type(8)))  __bf16 v8bf;
typedef __attribute__((ext_vector_type(4)))  __bf16 v4bf;
typedef __attribute__((ext_vector_type(8)))  float  v8f;
typedef __attribute__((ext_vector_type(4)))  int    v4i;

#define OUT_F 4096
#define IN_F  4096
#define M_TOK 2048
#define RANK  64

// ---------------------------------------------------------------------------
// Async Global->LDS (CDNA5 ASYNCcnt path) with portable fallback.
// Probe result: builtin exists; param0 is `int __vector(4) __device__*`
// (= v4i in address_space(1)). Param1 is the LDS destination.
// ---------------------------------------------------------------------------
#define AS1 __attribute__((address_space(1)))
#define AS3 __attribute__((address_space(3)))

#if __has_builtin(__builtin_amdgcn_global_load_async_to_lds_b128)
#define HAVE_ASYNC_LDS 1
#define COPY_B128_TO_LDS(l, g)                                  \
  __builtin_amdgcn_global_load_async_to_lds_b128(               \
      (AS1 v4i*)(g), (AS3 v4i*)(l), 0, 0)
#else
#define HAVE_ASYNC_LDS 0
#define COPY_B128_TO_LDS(l, g) (*(v8bf*)(l) = *(const v8bf*)(g))
#endif

__device__ __forceinline__ void wait_fills_and_sync() {
#if HAVE_ASYNC_LDS
#if __has_builtin(__builtin_amdgcn_s_wait_asynccnt)
  __builtin_amdgcn_s_wait_asynccnt(0);
#else
  asm volatile("s_wait_asynccnt 0x0" ::: "memory");
#endif
#endif
  __syncthreads();
}

__constant__ float NF4_TAB[16] = {
    -1.0f, -0.6961928009986877f, -0.5250730514526367f, -0.39491748809814453f,
    -0.28444138169288635f, -0.18477343022823334f, -0.09105003625154495f, 0.0f,
    0.07958029955625534f, 0.16093020141124725f, 0.24611230194568634f,
    0.33791524171829224f, 0.44070982933044434f, 0.5626170039176941f,
    0.7229568362236023f, 1.0f};

// ---------------------------------------------------------------------------
// 1) NF4 dequant, 4 packed int32 (= 8 weights) per thread -> one b128 store.
// ---------------------------------------------------------------------------
__global__ __launch_bounds__(256) void nf4_dequant_kernel(
    const int4* __restrict__ qw, const float* __restrict__ absmax,
    v8bf* __restrict__ W, int n_packed4) {
  int t = blockIdx.x * blockDim.x + threadIdx.x;
  if (t >= n_packed4) return;
  int4 q = qw[t];
  float s = absmax[t >> 3];  // (4t) >> 5
  int v[4] = {q.x, q.y, q.z, q.w};
  v8bf o;
#pragma unroll
  for (int j = 0; j < 4; ++j) {
    o[2 * j + 0] = (bf16)(NF4_TAB[v[j] & 15] * s);
    o[2 * j + 1] = (bf16)(NF4_TAB[(v[j] >> 4) & 15] * s);
  }
  W[t] = o;
}

// ---------------------------------------------------------------------------
// 2) f32 -> bf16 cast, x4 vectorized.
// ---------------------------------------------------------------------------
__global__ __launch_bounds__(256) void cast_f32_bf16_kernel(
    const float4* __restrict__ src, v4bf* __restrict__ dst, int n4) {
  int i = blockIdx.x * blockDim.x + threadIdx.x;
  if (i >= n4) return;
  float4 f = src[i];
  v4bf o;
  o[0] = (bf16)f.x; o[1] = (bf16)f.y; o[2] = (bf16)f.z; o[3] = (bf16)f.w;
  dst[i] = o;
}

// ---------------------------------------------------------------------------
// WMMA fragment loaders (wave32, EXEC all ones). Work on global or LDS ptrs.
// ---------------------------------------------------------------------------
union Frag {
  v16bf v;
  v8bf  h[2];
};

__device__ __forceinline__ v16bf load_frag_a_ptr(const bf16* p) {
  Frag f;
  f.h[0] = *(const v8bf*)(p);
  f.h[1] = *(const v8bf*)(p + 16);
  return f.v;
}

__device__ __forceinline__ v16bf load_frag_b_ptr(const bf16* p) {
  Frag f;
  f.h[0] = *(const v8bf*)(p);
  f.h[1] = *(const v8bf*)(p + 16);
  return f.v;
}

__device__ __forceinline__ v8f vzero8() {
  v8f z;
#pragma unroll
  for (int e = 0; e < 8; ++e) z[e] = 0.0f;
  return z;
}

#define WMMA_BF16(a, b, c)                                               \
  __builtin_amdgcn_wmma_f32_16x16x32_bf16(false, (a), false, (b),        \
                                          (short)0, (c), false, false)

// ---------------------------------------------------------------------------
// 3) LoRA stage 1: T[M_TOK][RANK] = X @ A^T (K = IN_F), bf16 out.
// ---------------------------------------------------------------------------
__global__ __launch_bounds__(256) void lora_stage1_kernel(
    const bf16* __restrict__ Xb,  // [M_TOK][IN_F]
    const bf16* __restrict__ Ab,  // [RANK][IN_F]
    bf16* __restrict__ Tb) {      // [M_TOK][RANK]
  const int lane = threadIdx.x & 31;
  const int wave = threadIdx.x >> 5;
  const int wm = wave >> 1, wn = wave & 1;
  const int m0 = blockIdx.x * 256 + wm * 64;
  const int n0 = wn * 32;
  const int r16 = lane & 15;
  const int ka = (lane >> 4) << 3;
  const int kb = (lane >> 4) << 3;

  v8f acc[4][2];
#pragma unroll
  for (int mi = 0; mi < 4; ++mi)
#pragma unroll
    for (int ni = 0; ni < 2; ++ni) acc[mi][ni] = vzero8();

  for (int k = 0; k < IN_F; k += 32) {
    v16bf a[4], b[2];
#pragma unroll
    for (int mi = 0; mi < 4; ++mi)
      a[mi] = load_frag_a_ptr(Xb + (size_t)(m0 + mi * 16 + r16) * IN_F + k + ka);
#pragma unroll
    for (int ni = 0; ni < 2; ++ni)
      b[ni] = load_frag_b_ptr(Ab + (size_t)(n0 + ni * 16 + r16) * IN_F + k + kb);
#pragma unroll
    for (int mi = 0; mi < 4; ++mi)
#pragma unroll
      for (int ni = 0; ni < 2; ++ni) acc[mi][ni] = WMMA_BF16(a[mi], b[ni], acc[mi][ni]);
  }

#pragma unroll
  for (int mi = 0; mi < 4; ++mi)
#pragma unroll
    for (int ni = 0; ni < 2; ++ni) {
      int row = m0 + mi * 16 + ((lane >> 4) << 3);
      int col = n0 + ni * 16 + r16;
#pragma unroll
      for (int r = 0; r < 8; ++r)
        Tb[(size_t)(row + r) * RANK + col] = (bf16)acc[mi][ni][r];
    }
}

// ---------------------------------------------------------------------------
// 4) Main fused GEMM with double-buffered async LDS staging.
//    WG: 8 waves = 4(M) x 2(N); wave tile 64x64; WG tile 256x128.
//    grid = (OUT_F/128, M_TOK/256) = (32, 8)
// ---------------------------------------------------------------------------
#define KSTEP 32
#define LDPAD 40  // 80-byte row pitch: 16B-aligned chunks, spreads LDS banks

__global__ __launch_bounds__(256) void qlora_gemm_kernel(
    const bf16* __restrict__ Xb,  // [M_TOK][IN_F]
    const bf16* __restrict__ Wb,  // [OUT_F][IN_F]
    const bf16* __restrict__ Tb,  // [M_TOK][RANK]
    const bf16* __restrict__ Bb,  // [OUT_F][RANK]
    float* __restrict__ out) {    // [M_TOK][OUT_F]
  __shared__ bf16 Xs[2][256][LDPAD];
  __shared__ bf16 Ws[2][128][LDPAD];

  const int tid = threadIdx.x;
  const int lane = tid & 31;
  const int wave = tid >> 5;
  const int wm = wave >> 1, wn = wave & 1;   // 4 x 2 wave grid
  const int m_base = blockIdx.y * 256;
  const int n_base = blockIdx.x * 128;
  const int r16 = lane & 15;
  const int ka = (lane >> 4) << 3;
  const int kb = (lane >> 4) << 3;

  const int wrow = tid >> 1;          // W fill: 2 threads per row of 128
  const int wko = (tid & 1) * 16;     // which 16-elem half of the k-step

  v8f acc[4][4];
#pragma unroll
  for (int mi = 0; mi < 4; ++mi)
#pragma unroll
    for (int ni = 0; ni < 4; ++ni) acc[mi][ni] = vzero8();

  // ---- fill buffer 0 for k = 0 ----
#pragma unroll
  for (int h = 0; h < 4; ++h)
    COPY_B128_TO_LDS(&Xs[0][tid][h * 8],
                     Xb + (size_t)(m_base + tid) * IN_F + h * 8);
#pragma unroll
  for (int h = 0; h < 2; ++h)
    COPY_B128_TO_LDS(&Ws[0][wrow][wko + h * 8],
                     Wb + (size_t)(n_base + wrow) * IN_F + wko + h * 8);

  const int NK = IN_F / KSTEP;
  for (int kt = 0; kt < NK; ++kt) {
    const int cur = kt & 1;
    wait_fills_and_sync();  // buffer `cur` ready; prior reads of `cur^1` done

    if (kt + 1 < NK) {
      const int nxt = cur ^ 1;
      const int kg = (kt + 1) * KSTEP;
#pragma unroll
      for (int h = 0; h < 4; ++h)
        COPY_B128_TO_LDS(&Xs[nxt][tid][h * 8],
                         Xb + (size_t)(m_base + tid) * IN_F + kg + h * 8);
#pragma unroll
      for (int h = 0; h < 2; ++h)
        COPY_B128_TO_LDS(&Ws[nxt][wrow][wko + h * 8],
                         Wb + (size_t)(n_base + wrow) * IN_F + kg + wko + h * 8);
    }

    v16bf a[4], b[4];
#pragma unroll
    for (int mi = 0; mi < 4; ++mi)
      a[mi] = load_frag_a_ptr(&Xs[cur][wm * 64 + mi * 16 + r16][ka]);
#pragma unroll
    for (int ni = 0; ni < 4; ++ni)
      b[ni] = load_frag_b_ptr(&Ws[cur][wn * 64 + ni * 16 + r16][kb]);
#pragma unroll
    for (int mi = 0; mi < 4; ++mi)
#pragma unroll
      for (int ni = 0; ni < 4; ++ni) acc[mi][ni] = WMMA_BF16(a[mi], b[ni], acc[mi][ni]);
  }

  // ---- LoRA tail: K = RANK = 64, direct from L2 (tiny) ----
#pragma unroll
  for (int k = 0; k < RANK; k += 32) {
    v16bf a[4], b[4];
#pragma unroll
    for (int mi = 0; mi < 4; ++mi)
      a[mi] = load_frag_a_ptr(Tb + (size_t)(m_base + wm * 64 + mi * 16 + r16) * RANK + k + ka);
#pragma unroll
    for (int ni = 0; ni < 4; ++ni)
      b[ni] = load_frag_b_ptr(Bb + (size_t)(n_base + wn * 64 + ni * 16 + r16) * RANK + k + kb);
#pragma unroll
    for (int mi = 0; mi < 4; ++mi)
#pragma unroll
      for (int ni = 0; ni < 4; ++ni) acc[mi][ni] = WMMA_BF16(a[mi], b[ni], acc[mi][ni]);
  }

  // ---- store f32 ----
#pragma unroll
  for (int mi = 0; mi < 4; ++mi)
#pragma unroll
    for (int ni = 0; ni < 4; ++ni) {
      int row = m_base + wm * 64 + mi * 16 + ((lane >> 4) << 3);
      int col = n_base + wn * 64 + ni * 16 + r16;
#pragma unroll
      for (int r = 0; r < 8; ++r)
        out[(size_t)(row + r) * OUT_F + col] = acc[mi][ni][r];
    }
}

// ---------------------------------------------------------------------------
// Host launcher
// ---------------------------------------------------------------------------
extern "C" void kernel_launch(void* const* d_in, const int* in_sizes, int n_in,
                              void* d_out, int out_size, void* d_ws,
                              size_t ws_size, hipStream_t stream) {
  const float* input   = (const float*)d_in[0];  // [2,1024,4096] f32
  const int*   qweight = (const int*)d_in[1];    // [8M,1] int32 (1 byte each)
  const float* absmax  = (const float*)d_in[2];  // [256K,1] f32
  const float* lora_A  = (const float*)d_in[3];  // [64,4096] f32
  const float* lora_B  = (const float*)d_in[4];  // [4096,64] f32
  float* out = (float*)d_out;

  char* ws = (char*)d_ws;
  size_t offW = 0;                                   // W bf16: 32 MB
  size_t offX = offW + (size_t)OUT_F * IN_F * 2;     // X bf16: 16 MB
  size_t offA = offX + (size_t)M_TOK * IN_F * 2;     // A bf16: 512 KB
  size_t offB = offA + (size_t)RANK * IN_F * 2;      // B bf16: 512 KB
  size_t offT = offB + (size_t)OUT_F * RANK * 2;     // T bf16: 256 KB
  size_t need = offT + (size_t)M_TOK * RANK * 2;
  if (ws_size < need) return;

  bf16* Wb = (bf16*)(ws + offW);
  bf16* Xb = (bf16*)(ws + offX);
  bf16* Ab = (bf16*)(ws + offA);
  bf16* Bb = (bf16*)(ws + offB);
  bf16* Tb = (bf16*)(ws + offT);

  const int n_packed4 = OUT_F * IN_F / 8;  // 4 packed int32 per thread

  nf4_dequant_kernel<<<n_packed4 / 256, 256, 0, stream>>>(
      (const int4*)qweight, absmax, (v8bf*)Wb, n_packed4);
  cast_f32_bf16_kernel<<<(M_TOK * IN_F / 4) / 256, 256, 0, stream>>>(
      (const float4*)input, (v4bf*)Xb, M_TOK * IN_F / 4);
  cast_f32_bf16_kernel<<<(RANK * IN_F / 4) / 256, 256, 0, stream>>>(
      (const float4*)lora_A, (v4bf*)Ab, RANK * IN_F / 4);
  cast_f32_bf16_kernel<<<(OUT_F * RANK / 4) / 256, 256, 0, stream>>>(
      (const float4*)lora_B, (v4bf*)Bb, OUT_F * RANK / 4);

  lora_stage1_kernel<<<M_TOK / 256, 256, 0, stream>>>(Xb, Ab, Tb);

  dim3 grid(OUT_F / 128, M_TOK / 256);  // (32, 8)
  qlora_gemm_kernel<<<grid, 256, 0, stream>>>(Xb, Wb, Tb, Bb, out);
}
